// RNNActor_55327768707181
// MI455X (gfx1250) — compile-verified, matrix operand order checked by reading, not run on hardware
//
#include <hip/hip_runtime.h>

// RNN actor on MI455X (gfx1250), wave32 + WMMA.
//   K1 encode:  feat = tanh(obs @ W_enc + b_enc)          -> f16 ws (parallel)
//   K2 gates_x: gx   = feat @ W_ih^T + (b_ih + b_hh)      -> f16 ws (parallel, 137 GFLOP)
//   K3 scan:    512 independent row recurrences, 32 WGs x 16 rows; W_hh LDS-resident;
//               gate-column ownership chosen so the cell update is lane-local registers.
// ws needs 2 * T*B*512 f16 = 512 MB.

#define T_DIM  512
#define B_DIM  512
#define D_IN   128
#define D_HID  512
#define D_RNN  128
#define D_ACT  16
#define G_DIM  512            // 4 * D_RNN
#define LOG_2PI 1.8378770664093453f

#define LP_OFF  ((size_t)T_DIM * B_DIM * D_ACT)            // 4194304
#define HT_OFF  (LP_OFF + (size_t)T_DIM * B_DIM)           // 4456448
#define CT_OFF  (HT_OFF + (size_t)B_DIM * D_RNN)           // 4521984

typedef __attribute__((ext_vector_type(16))) _Float16 v16h;
typedef __attribute__((ext_vector_type(8)))  float    v8f;
typedef __attribute__((ext_vector_type(4)))  _Float16 h4;

__device__ __forceinline__ v8f vzero8() {
  v8f z;
#pragma unroll
  for (int i = 0; i < 8; ++i) z[i] = 0.0f;
  return z;
}

// 16x32 half fragment in CDNA5 WMMA A/B register layout from an LDS tile
// stored row-major ([M][K] for A, transposed [N][K] for B), `stride` halves.
// Lane 0-15: row=lane,    halves 0..7 = K k0+0..7,  halves 8..15 = K k0+16..23
// Lane16-31: row=lane-16, halves 0..7 = K k0+8..15, halves 8..15 = K k0+24..31
__device__ __forceinline__ v16h load_frag16x32(const _Float16* s, int stride,
                                               int row0, int k0, unsigned lane) {
  const int r  = lane & 15;
  const int kb = (lane & 16) ? 8 : 0;
  const _Float16* p = s + (row0 + r) * stride + k0 + kb;
  v16h f;
#pragma unroll
  for (int i = 0; i < 8; ++i) { f[i] = p[i]; f[8 + i] = p[16 + i]; }
  return f;
}

__device__ __forceinline__ h4 cvt4(float4 f) {
  h4 p;
  p[0] = (_Float16)f.x; p[1] = (_Float16)f.y;
  p[2] = (_Float16)f.z; p[3] = (_Float16)f.w;
  return p;
}

__device__ __forceinline__ float sigmf(float x) {
  return 1.0f / (1.0f + __expf(-x));
}

// ---------------------------------------------------------------------------
// K1: feat[M,512] = tanh(obs[M,128] @ W_enc[128,512] + b_enc)   (f16 out)
// ---------------------------------------------------------------------------
__global__ __launch_bounds__(256) void k_encode(const float* __restrict__ obs,
                                                const float* __restrict__ W_enc,
                                                const float* __restrict__ b_enc,
                                                _Float16* __restrict__ feat) {
  __shared__ _Float16 sA[128 * 40];   // [m][k] padded (80B rows: b128-aligned frags)
  __shared__ _Float16 sB[128 * 40];   // [n][k] padded (transposed stage)
  const int tid = threadIdx.x;
  const unsigned lane = tid & 31u;
  const int wid = tid >> 5;
  const int nb = (blockIdx.x & 3) * 128;
  const int mb = (blockIdx.x >> 2) * 128;
  const int m0w = (wid & 3) * 32;
  const int n0w = (wid >> 2) * 64;

  v8f acc[2][4];
#pragma unroll
  for (int i = 0; i < 2; ++i)
#pragma unroll
    for (int j = 0; j < 4; ++j) acc[i][j] = vzero8();

  for (int kt = 0; kt < D_IN / 32; ++kt) {
    // A tile: b128 global loads, packed f16, b64 LDS stores
    for (int idx = tid; idx < 128 * 8; idx += 256) {
      int m = idx >> 3, k4 = (idx & 7) * 4;
      float4 f = *(const float4*)&obs[(size_t)(mb + m) * D_IN + kt * 32 + k4];
      *(h4*)&sA[m * 40 + k4] = cvt4(f);
    }
    // B tile: b128 global loads over n (coalesced), transpose-scatter to LDS
    for (int idx = tid; idx < 32 * 32; idx += 256) {
      int k = idx >> 5, n4 = (idx & 31) * 4;
      float4 f = *(const float4*)&W_enc[(size_t)(kt * 32 + k) * D_HID + nb + n4];
      sB[(n4 + 0) * 40 + k] = (_Float16)f.x;
      sB[(n4 + 1) * 40 + k] = (_Float16)f.y;
      sB[(n4 + 2) * 40 + k] = (_Float16)f.z;
      sB[(n4 + 3) * 40 + k] = (_Float16)f.w;
    }
    __syncthreads();
    v16h af[2], bf[4];
#pragma unroll
    for (int i = 0; i < 2; ++i) af[i] = load_frag16x32(sA, 40, m0w + i * 16, 0, lane);
#pragma unroll
    for (int j = 0; j < 4; ++j) bf[j] = load_frag16x32(sB, 40, n0w + j * 16, 0, lane);
#pragma unroll
    for (int i = 0; i < 2; ++i)
#pragma unroll
      for (int j = 0; j < 4; ++j)
        acc[i][j] = __builtin_amdgcn_wmma_f32_16x16x32_f16(
            false, af[i], false, bf[j], (short)0, acc[i][j], false, false);
    __syncthreads();
  }

  const int nlo  = lane & 15;
  const int roff = (lane & 16) ? 8 : 0;
#pragma unroll
  for (int i = 0; i < 2; ++i)
#pragma unroll
    for (int j = 0; j < 4; ++j) {
      const int n = nb + n0w + j * 16 + nlo;
      const float bias = b_enc[n];
#pragma unroll
      for (int v = 0; v < 8; ++v) {
        const int m = mb + m0w + i * 16 + v + roff;
        feat[(size_t)m * D_HID + n] = (_Float16)tanhf(acc[i][j][v] + bias);
      }
    }
}

// ---------------------------------------------------------------------------
// K2: gx[M,512] = feat[M,512] @ W_ih^T[512,512] + (b_ih + b_hh)  (f16 out)
// ---------------------------------------------------------------------------
__global__ __launch_bounds__(256) void k_gate(const _Float16* __restrict__ feat,
                                              const float* __restrict__ W_ih,
                                              const float* __restrict__ b_ih,
                                              const float* __restrict__ b_hh,
                                              _Float16* __restrict__ gx) {
  __shared__ _Float16 sA[128 * 40];
  __shared__ _Float16 sB[128 * 40];   // [n][k]; B[k][n] = W_ih[n][k]
  const int tid = threadIdx.x;
  const unsigned lane = tid & 31u;
  const int wid = tid >> 5;
  const int nb = (blockIdx.x & 3) * 128;
  const int mb = (blockIdx.x >> 2) * 128;
  const int m0w = (wid & 3) * 32;
  const int n0w = (wid >> 2) * 64;

  v8f acc[2][4];
#pragma unroll
  for (int i = 0; i < 2; ++i)
#pragma unroll
    for (int j = 0; j < 4; ++j) acc[i][j] = vzero8();

  for (int kt = 0; kt < D_HID / 32; ++kt) {
    // A tile already f16: straight b128 copies
    for (int idx = tid; idx < 128 * 4; idx += 256) {
      int m = idx >> 2, k8 = (idx & 3) * 8;
      *(uint4*)&sA[m * 40 + k8] =
          *(const uint4*)&feat[(size_t)(mb + m) * D_HID + kt * 32 + k8];
    }
    // W_ih rows: b128 global over k, packed f16, b64 LDS stores
    for (int idx = tid; idx < 128 * 8; idx += 256) {
      int n = idx >> 3, k4 = (idx & 7) * 4;
      float4 f = *(const float4*)&W_ih[(size_t)(nb + n) * D_HID + kt * 32 + k4];
      *(h4*)&sB[n * 40 + k4] = cvt4(f);
    }
    __syncthreads();
    v16h af[2], bf[4];
#pragma unroll
    for (int i = 0; i < 2; ++i) af[i] = load_frag16x32(sA, 40, m0w + i * 16, 0, lane);
#pragma unroll
    for (int j = 0; j < 4; ++j) bf[j] = load_frag16x32(sB, 40, n0w + j * 16, 0, lane);
#pragma unroll
    for (int i = 0; i < 2; ++i)
#pragma unroll
      for (int j = 0; j < 4; ++j)
        acc[i][j] = __builtin_amdgcn_wmma_f32_16x16x32_f16(
            false, af[i], false, bf[j], (short)0, acc[i][j], false, false);
    __syncthreads();
  }

  const int nlo  = lane & 15;
  const int roff = (lane & 16) ? 8 : 0;
#pragma unroll
  for (int i = 0; i < 2; ++i)
#pragma unroll
    for (int j = 0; j < 4; ++j) {
      const int n = nb + n0w + j * 16 + nlo;
      const float bias = b_ih[n] + b_hh[n];
#pragma unroll
      for (int v = 0; v < 8; ++v) {
        const int m = mb + m0w + i * 16 + v + roff;
        gx[(size_t)m * G_DIM + n] = (_Float16)(acc[i][j][v] + bias);
      }
    }
}

// ---------------------------------------------------------------------------
// K3: scan. 32 WGs x 16 batch rows. Wave w owns gate columns
// {j, 128+j, 256+j, 384+j : j in [w*16, w*16+16)}, so i/f/g/o of one cell sit
// in the same lane & accumulator slot of the 4 WMMA fragments -> register-local
// LSTM update, no gate LDS roundtrip.
// ---------------------------------------------------------------------------
#define WHH_STR 136   // 272B rows: 16B-aligned ds_load_b128, bank stride 4 -> no conflicts

__global__ __launch_bounds__(256) void k_scan(const _Float16* __restrict__ gx,
                                              const float* __restrict__ done,
                                              const float* __restrict__ h0,
                                              const float* __restrict__ c0,
                                              const float* __restrict__ W_hh,
                                              const float* __restrict__ W_mean,
                                              const float* __restrict__ b_mean,
                                              const float* __restrict__ log_std,
                                              float* __restrict__ out) {
  __shared__ _Float16 sWhh[G_DIM * WHH_STR];  // [n][k], B[k][n] = W_hh[n][k]  (136 KB)
  __shared__ _Float16 sWm[D_ACT * WHH_STR];   // [n][k], B[k][n] = W_mean[k][n]
  __shared__ _Float16 sH[16 * WHH_STR];       // current h rows (A matrix)
  __shared__ float    sDone[16];              // 1 - done mask, staged one step ahead

  const int tid = threadIdx.x;
  const unsigned lane = tid & 31u;
  const int wid = tid >> 5;
  const int brow0 = blockIdx.x * 16;
  const int nlo = lane & 15;
  const int roff = (lane & 16) ? 8 : 0;       // owned rows: b_local = v + roff
  const int j0 = wid * 16 + nlo;              // owned hidden column

  // stage W_hh (vectorized) and W_mean^T
  for (int idx = tid; idx < (G_DIM * D_RNN) / 4; idx += 256) {
    int n = idx >> 5, k4 = (idx & 31) * 4;
    float4 f = *(const float4*)&W_hh[(size_t)n * D_RNN + k4];
    *(h4*)&sWhh[n * WHH_STR + k4] = cvt4(f);
  }
  for (int idx = tid; idx < D_RNN * D_ACT; idx += 256) {
    int k = idx >> 4, n = idx & 15;
    sWm[n * WHH_STR + k] = (_Float16)W_mean[idx];   // idx == k*16+n
  }
  if (tid < 16) sDone[tid] = 1.0f - done[brow0 + tid];   // mask for t = 0

  // registers: cell (b = v + roff, j = j0) for v = 0..7
  float hreg[8], creg[8];
#pragma unroll
  for (int v = 0; v < 8; ++v) {
    hreg[v] = h0[(size_t)(brow0 + v + roff) * D_RNN + j0];
    creg[v] = c0[(size_t)(brow0 + v + roff) * D_RNN + j0];
  }
  float lp = 0.0f;
#pragma unroll
  for (int a = 0; a < D_ACT; ++a) lp += -log_std[a] - 0.5f * LOG_2PI;
  __syncthreads();

  for (int t = 0; t < T_DIM; ++t) {
    // -- P1: done-mask resets state, publish masked h
#pragma unroll
    for (int v = 0; v < 8; ++v) {
      const float m = sDone[v + roff];
      creg[v] *= m;
      hreg[v] *= m;
      sH[(v + roff) * WHH_STR + j0] = (_Float16)hreg[v];
    }
    if (t + 1 < T_DIM)  // prefetch next step's gx block (16KB across 256 threads)
      __builtin_prefetch(gx + ((size_t)(t + 1) * B_DIM + brow0) * G_DIM +
                         (size_t)tid * 32, 0, 0);
    __syncthreads();

    // -- P2: gates = gx[t] + h @ W_hh^T ; wave w's 4 tiles at n = g*128 + w*16
    v8f acc[4];
    {
      const size_t rowbase = (size_t)t * B_DIM + brow0;
#pragma unroll
      for (int g = 0; g < 4; ++g) {
        const int n = g * 128 + j0;
#pragma unroll
        for (int v = 0; v < 8; ++v)
          acc[g][v] = (float)gx[(rowbase + v + roff) * G_DIM + n];
      }
#pragma unroll
      for (int kc = 0; kc < D_RNN / 32; ++kc) {
        const v16h a = load_frag16x32(sH, WHH_STR, 0, kc * 32, lane);
#pragma unroll
        for (int g = 0; g < 4; ++g) {
          const v16h bfr =
              load_frag16x32(sWhh, WHH_STR, g * 128 + wid * 16, kc * 32, lane);
          acc[g] = __builtin_amdgcn_wmma_f32_16x16x32_f16(
              false, a, false, bfr, (short)0, acc[g], false, false);
        }
      }
    }
    __syncthreads();   // all sH reads done before overwrite

    // -- P3: lane-local LSTM cell update (torch i,f,g,o), publish new h
#pragma unroll
    for (int v = 0; v < 8; ++v) {
      const float ig = sigmf(acc[0][v]);
      const float fg = sigmf(acc[1][v]);
      const float gg = tanhf(acc[2][v]);
      const float og = sigmf(acc[3][v]);
      const float c = fg * creg[v] + ig * gg;
      const float h = og * tanhf(c);
      creg[v] = c;
      hreg[v] = h;
      sH[(v + roff) * WHH_STR + j0] = (_Float16)h;
    }
    if (tid < 16)      // stage mask for t+1
      sDone[tid] = (t + 1 < T_DIM)
                 ? 1.0f - done[(size_t)(t + 1) * B_DIM + brow0 + tid] : 1.0f;
    __syncthreads();

    // -- P4: mean head (wave 0) + constant log_prob
    if (wid == 0) {
      v8f macc;
#pragma unroll
      for (int v = 0; v < 8; ++v) macc[v] = b_mean[nlo];
#pragma unroll
      for (int kc = 0; kc < D_RNN / 32; ++kc) {
        const v16h a = load_frag16x32(sH, WHH_STR, 0, kc * 32, lane);
        const v16h bfr = load_frag16x32(sWm, WHH_STR, 0, kc * 32, lane);
        macc = __builtin_amdgcn_wmma_f32_16x16x32_f16(
            false, a, false, bfr, (short)0, macc, false, false);
      }
      const size_t ob = (size_t)t * B_DIM + brow0;
#pragma unroll
      for (int v = 0; v < 8; ++v)
        out[(ob + v + roff) * D_ACT + nlo] = macc[v];
    }
    if (tid < 16)
      out[LP_OFF + (size_t)t * B_DIM + brow0 + tid] = lp;
    __syncthreads();
  }

  // final states hT / cT
#pragma unroll
  for (int v = 0; v < 8; ++v) {
    out[HT_OFF + (size_t)(brow0 + v + roff) * D_RNN + j0] = hreg[v];
    out[CT_OFF + (size_t)(brow0 + v + roff) * D_RNN + j0] = creg[v];
  }
}

// ---------------------------------------------------------------------------
extern "C" void kernel_launch(void* const* d_in, const int* in_sizes, int n_in,
                              void* d_out, int out_size, void* d_ws, size_t ws_size,
                              hipStream_t stream) {
  const float* obs     = (const float*)d_in[0];
  const float* h0      = (const float*)d_in[1];
  const float* c0      = (const float*)d_in[2];
  const float* done    = (const float*)d_in[3];
  const float* W_enc   = (const float*)d_in[4];
  const float* b_enc   = (const float*)d_in[5];
  const float* W_ih    = (const float*)d_in[6];
  const float* W_hh    = (const float*)d_in[7];
  const float* b_ih    = (const float*)d_in[8];
  const float* b_hh    = (const float*)d_in[9];
  const float* W_mean  = (const float*)d_in[10];
  const float* b_mean  = (const float*)d_in[11];
  const float* log_std = (const float*)d_in[12];
  float* out = (float*)d_out;

  _Float16* feat = (_Float16*)d_ws;                          // [T*B, 512] f16
  _Float16* gx   = feat + (size_t)T_DIM * B_DIM * D_HID;     // [T*B, 512] f16

  const int mtiles = (T_DIM * B_DIM) / 128;                  // 2048
  k_encode<<<dim3(mtiles * (D_HID / 128)), dim3(256), 0, stream>>>(obs, W_enc, b_enc, feat);
  k_gate  <<<dim3(mtiles * (G_DIM / 128)), dim3(256), 0, stream>>>(feat, W_ih, b_ih, b_hh, gx);
  k_scan  <<<dim3(B_DIM / 16), dim3(256), 0, stream>>>(gx, done, h0, c0, W_hh,
                                                       W_mean, b_mean, log_std, out);
}
